// InteractionModule_23725399343496
// MI455X (gfx1250) — matrix operand order repaired and checked by reading, not compiled, not run on hardware
//
#include <hip/hip_runtime.h>
#include <hip/hip_bf16.h>

typedef __attribute__((ext_vector_type(16))) _Float16 v16h;
typedef __attribute__((ext_vector_type(8)))  float    v8f;

#define N_ROWS 4096
#define HDIM   64

// ---------------------------------------------------------------------------
// Kernel 0: front-end GEMMs (x = in@We+be; q = x@Wq+bq; k = x@Wk+bk),
// emitting q/k in f16 *pre-swizzled* into the CDNA5 WMMA A/B fragment layouts
// (ISA 7.12.2, 16-bit A 16x32 and B 32x16, wave32) so the WMMA kernel can do
// contiguous 32B-per-lane fragment loads.
// One thread per row; weights broadcast (uniform addresses -> scalar loads).
// ---------------------------------------------------------------------------
__global__ void __launch_bounds__(256)
k_front(const float* __restrict__ in, const float* __restrict__ We,
        const float* __restrict__ be, const float* __restrict__ Wq,
        const float* __restrict__ bq, const float* __restrict__ Wk,
        const float* __restrict__ bk,
        _Float16* __restrict__ qsw, _Float16* __restrict__ ksw) {
  int i = blockIdx.x * blockDim.x + threadIdx.x;
  if (i >= N_ROWS) return;

  float xin[8];
#pragma unroll
  for (int d = 0; d < 8; ++d) xin[d] = in[i * 8 + d];

  float x[HDIM];
  for (int h = 0; h < HDIM; ++h) {
    float acc = be[h];
#pragma unroll
    for (int d = 0; d < 8; ++d) acc += xin[d] * We[d * HDIM + h];
    x[h] = acc;
  }

  const int tile = i >> 4;   // 16-row tile
  const int m    = i & 15;   // row-in-tile

  for (int h = 0; h < HDIM; ++h) {
    float accq = bq[h], acck = bk[h];
#pragma unroll
    for (int h2 = 0; h2 < HDIM; ++h2) {
      accq += x[h2] * Wq[h2 * HDIM + h];
      acck += x[h2] * Wk[h2 * HDIM + h];
    }
    const int g  = h >> 5;     // which of two K=32 fragments
    const int Kr = h & 31;     // K within fragment

    // A-matrix layout (16-bit A 16x32): lanes 0-15 hold K {0..7,16..23},
    // lanes 16-31 hold K {8..15,24..31}, packed as v16h element j.
    int laneA, jA;
    if      (Kr <  8) { laneA = m;      jA = Kr;      }
    else if (Kr < 16) { laneA = m + 16; jA = Kr - 8;  }
    else if (Kr < 24) { laneA = m;      jA = Kr - 8;  }
    else              { laneA = m + 16; jA = Kr - 16; }
    qsw[((size_t)(tile * 2 + g) * 32 + laneA) * 16 + jA] = (_Float16)accq;

    // B-matrix layout (16-bit B 32x16): same K order as A (lanes 0-15 hold
    // K {0..7,16..23}, lanes 16-31 hold K {8..15,24..31}); lane%16 = N.
    int laneB = ((Kr >> 3) & 1) ? (m + 16) : m;
    int jB    = (Kr & 7) + ((Kr >> 4) << 3);
    ksw[((size_t)(tile * 2 + g) * 32 + laneB) * 16 + jB] = (_Float16)acck;
  }
}

// ---------------------------------------------------------------------------
// Kernel 1: scores = exp(q@k^T * 1/8) via v_wmma_f32_16x16x32_f16, plus
// per-row sums (for softmax) via intra-wave shuffle reduce + one atomicAdd.
// One block per 16-row stripe; 8 waves each cover 32 of the 256 column tiles.
// Normalization by 1/rowsum is deferred to the consumers (saves a full
// 128MB pass over the map).
// ---------------------------------------------------------------------------
__global__ void __launch_bounds__(256)
k_scores(const _Float16* __restrict__ qsw, const _Float16* __restrict__ ksw,
         float* __restrict__ scores, float* __restrict__ rowsum) {
  const int rowtile = blockIdx.x;          // 0..255
  const int wave    = threadIdx.x >> 5;    // 0..7
  const int lane    = threadIdx.x & 31;

  const v16h* qv = (const v16h*)qsw;
  const v16h* kv = (const v16h*)ksw;

  const v16h a0 = qv[(size_t)(rowtile * 2 + 0) * 32 + lane];
  const v16h a1 = qv[(size_t)(rowtile * 2 + 1) * 32 + lane];

  const int row0 = rowtile * 16;
  const int nsub = lane & 15;
  const int moff = (lane < 16) ? 0 : 8;
  const float scale = 0.125f;   // 1/sqrt(64)

  float racc[8];
#pragma unroll
  for (int r = 0; r < 8; ++r) racc[r] = 0.f;

  for (int ct = wave * 32; ct < wave * 32 + 32; ++ct) {
    const v16h b0 = kv[(size_t)(ct * 2 + 0) * 32 + lane];
    const v16h b1 = kv[(size_t)(ct * 2 + 1) * 32 + lane];
    v8f c = {};
    c = __builtin_amdgcn_wmma_f32_16x16x32_f16(false, a0, false, b0,
                                               (short)0, c, false, false);
    c = __builtin_amdgcn_wmma_f32_16x16x32_f16(false, a1, false, b1,
                                               (short)0, c, false, false);
    const int col = ct * 16 + nsub;
#pragma unroll
    for (int r = 0; r < 8; ++r) {
      // C/D layout: VGPR r holds M = r (lanes 0-15) / r+8 (lanes 16-31).
      float e = __expf(c[r] * scale);
      scores[(size_t)(row0 + r + moff) * N_ROWS + col] = e;
      racc[r] += e;
    }
  }

  // Reduce each row-partial across the 16 lanes of its half-group.
#pragma unroll
  for (int r = 0; r < 8; ++r) {
    float v = racc[r];
    for (int off = 8; off; off >>= 1) v += __shfl_xor(v, off, 32);
    if (nsub == 0) atomicAdd(&rowsum[row0 + r + moff], v);
  }
}

// ---------------------------------------------------------------------------
// Kernel 2: one 5-point stencil iteration:
//   out = prelu(conv_row(cf,k1,b1) + conv_col(cf,k2,b2), a)
// float4 per thread; zero padding at edges. First iteration folds in the
// softmax normalization (value of cf0[r][c] = exp[r][c] / rowsum[r]).
// Working set (two 64MB buffers) fits in the 192MB L2.
// ---------------------------------------------------------------------------
__global__ void __launch_bounds__(256)
k_conv(const float* __restrict__ src, float* __restrict__ dst,
       const float* __restrict__ rowsum, int normalize,
       const float* __restrict__ k1, const float* __restrict__ b1p,
       const float* __restrict__ k2, const float* __restrict__ b2p,
       const float* __restrict__ pa) {
  const int idx = blockIdx.x * blockDim.x + threadIdx.x;
  const int r   = idx >> 10;          // row 0..4095
  const int c0  = (idx & 1023) << 2;  // first of 4 columns
  if (r >= N_ROWS) return;

  const size_t base = (size_t)r * N_ROWS;
  float4 C = *(const float4*)(src + base + c0);
  float4 U = make_float4(0.f, 0.f, 0.f, 0.f);
  float4 D = make_float4(0.f, 0.f, 0.f, 0.f);
  if (r > 0)           U = *(const float4*)(src + base - N_ROWS + c0);
  if (r < N_ROWS - 1)  D = *(const float4*)(src + base + N_ROWS + c0);
  float L = (c0 > 0)            ? src[base + c0 - 1] : 0.f;
  float R = (c0 + 4 < N_ROWS)   ? src[base + c0 + 4] : 0.f;

  if (normalize) {
    const float ic = 1.f / rowsum[r];
    C.x *= ic; C.y *= ic; C.z *= ic; C.w *= ic; L *= ic; R *= ic;
    if (r > 0)          { const float iu = 1.f / rowsum[r - 1];
                          U.x *= iu; U.y *= iu; U.z *= iu; U.w *= iu; }
    if (r < N_ROWS - 1) { const float id = 1.f / rowsum[r + 1];
                          D.x *= id; D.y *= id; D.z *= id; D.w *= id; }
  }

  const float k10 = k1[0], k12 = k1[2];
  const float k20 = k2[0], k22 = k2[2];
  const float cc  = k1[1] + k2[1];
  const float b   = b1p[0] + b2p[0];
  const float a   = pa[0];

  float4 o;
  o.x = k10 * L   + cc * C.x + k12 * C.y + k20 * U.x + k22 * D.x + b;
  o.y = k10 * C.x + cc * C.y + k12 * C.z + k20 * U.y + k22 * D.y + b;
  o.z = k10 * C.y + cc * C.z + k12 * C.w + k20 * U.z + k22 * D.z + b;
  o.w = k10 * C.z + cc * C.w + k12 * R   + k20 * U.w + k22 * D.w + b;
  o.x = o.x > 0.f ? o.x : a * o.x;
  o.y = o.y > 0.f ? o.y : a * o.y;
  o.z = o.z > 0.f ? o.z : a * o.z;
  o.w = o.w > 0.f ? o.w : a * o.w;
  *(float4*)(dst + base + c0) = o;
}

// ---------------------------------------------------------------------------
// Kernel 3: g_pre = (mask * softmax_scores) @ o, one wave per row.
// mask = (sigmoid(cf)>0.5) == (cf>0). o = inputs[:, :4] read in place.
// Softmax normalization (1/rowsum) applied once per row at the end.
// ---------------------------------------------------------------------------
__global__ void __launch_bounds__(256)
k_adjo(const float* __restrict__ cf, const float* __restrict__ se,
       const float* __restrict__ rowsum, const float* __restrict__ in,
       float* __restrict__ g_pre) {
  const int row  = (blockIdx.x * blockDim.x + threadIdx.x) >> 5;
  const int lane = threadIdx.x & 31;
  if (row >= N_ROWS) return;

  const size_t base = (size_t)row * N_ROWS;
  float a0 = 0.f, a1 = 0.f, a2 = 0.f, a3 = 0.f;
  for (int j = lane; j < N_ROWS; j += 32) {
    float v = cf[base + j];
    if (v > 0.f) {
      float s = se[base + j];
      float4 o4 = *(const float4*)(in + (size_t)j * 8);
      a0 += s * o4.x; a1 += s * o4.y; a2 += s * o4.z; a3 += s * o4.w;
    }
  }
  for (int off = 16; off; off >>= 1) {
    a0 += __shfl_xor(a0, off, 32);
    a1 += __shfl_xor(a1, off, 32);
    a2 += __shfl_xor(a2, off, 32);
    a3 += __shfl_xor(a3, off, 32);
  }
  if (lane == 0) {
    const float inv = 1.f / rowsum[row];
    float4 o; o.x = a0 * inv; o.y = a1 * inv; o.z = a2 * inv; o.w = a3 * inv;
    *(float4*)(g_pre + (size_t)row * 4) = o;
  }
}

// ---------------------------------------------------------------------------
// Kernel 4: tail MLP per row:
//   g = prelu(g_pre@Wg + bg);  pred = relu(g@Wp1+bp1)@Wp2 + bp2
// One thread per row; weights (<10KB) broadcast through L2.
// ---------------------------------------------------------------------------
__global__ void __launch_bounds__(256)
k_tail(const float* __restrict__ g_pre, const float* __restrict__ Wg,
       const float* __restrict__ bg, const float* __restrict__ ga,
       const float* __restrict__ Wp1, const float* __restrict__ bp1,
       const float* __restrict__ Wp2, const float* __restrict__ bp2,
       float* __restrict__ out) {
  const int i = blockIdx.x * blockDim.x + threadIdx.x;
  if (i >= N_ROWS) return;

  const float4 t = *(const float4*)(g_pre + (size_t)i * 4);
  const float a  = ga[0];

  float h1[32];
#pragma unroll
  for (int p = 0; p < 32; ++p) h1[p] = bp1[p];

  for (int h = 0; h < HDIM; ++h) {
    float gh = t.x * Wg[0 * HDIM + h] + t.y * Wg[1 * HDIM + h] +
               t.z * Wg[2 * HDIM + h] + t.w * Wg[3 * HDIM + h] + bg[h];
    gh = gh > 0.f ? gh : a * gh;
#pragma unroll
    for (int p = 0; p < 32; ++p) h1[p] += gh * Wp1[h * 32 + p];
  }

  float o0 = bp2[0], o1 = bp2[1], o2 = bp2[2], o3 = bp2[3];
#pragma unroll
  for (int p = 0; p < 32; ++p) {
    float hp = h1[p] > 0.f ? h1[p] : 0.f;
    o0 += hp * Wp2[p * 4 + 0];
    o1 += hp * Wp2[p * 4 + 1];
    o2 += hp * Wp2[p * 4 + 2];
    o3 += hp * Wp2[p * 4 + 3];
  }
  float4 r; r.x = o0; r.y = o1; r.z = o2; r.w = o3;
  *(float4*)(out + (size_t)i * 4) = r;
}

// ---------------------------------------------------------------------------
extern "C" void kernel_launch(void* const* d_in, const int* in_sizes, int n_in,
                              void* d_out, int out_size, void* d_ws,
                              size_t ws_size, hipStream_t stream) {
  (void)in_sizes; (void)n_in; (void)out_size; (void)ws_size;

  const float* inp     = (const float*)d_in[0];   // (1,4096,8)
  const float* We      = (const float*)d_in[1];
  const float* be      = (const float*)d_in[2];
  const float* Wq      = (const float*)d_in[3];
  const float* bq      = (const float*)d_in[4];
  const float* Wk      = (const float*)d_in[5];
  const float* bk      = (const float*)d_in[6];
  const float* prelu_a = (const float*)d_in[7];
  const float* conv1_k = (const float*)d_in[8];
  const float* conv1_b = (const float*)d_in[9];
  const float* conv2_k = (const float*)d_in[10];
  const float* conv2_b = (const float*)d_in[11];
  const float* Wg      = (const float*)d_in[12];
  const float* bg      = (const float*)d_in[13];
  const float* gcn_a   = (const float*)d_in[14];
  const float* Wp1     = (const float*)d_in[15];
  const float* bp1     = (const float*)d_in[16];
  const float* Wp2     = (const float*)d_in[17];
  const float* bp2     = (const float*)d_in[18];

  const size_t MAP = (size_t)N_ROWS * N_ROWS * sizeof(float);  // 64 MB
  char* ws = (char*)d_ws;
  float*    scores = (float*)(ws);               // exp(logits), 64 MB
  float*    buf0   = (float*)(ws + MAP);         // conv ping,   64 MB
  float*    buf1   = (float*)(ws + 2 * MAP);     // conv pong,   64 MB
  _Float16* qsw    = (_Float16*)(ws + 3 * MAP);  // 512 KB
  _Float16* ksw    = qsw + (size_t)N_ROWS * HDIM;
  float*    rowsum = (float*)(ksw + (size_t)N_ROWS * HDIM);
  float*    g_pre  = rowsum + N_ROWS;

  hipMemsetAsync(rowsum, 0, N_ROWS * sizeof(float), stream);

  k_front<<<N_ROWS / 256, 256, 0, stream>>>(inp, We, be, Wq, bq, Wk, bk,
                                            qsw, ksw);

  k_scores<<<N_ROWS / 16, 256, 0, stream>>>(qsw, ksw, scores, rowsum);

  const float* csrc = scores;
  float*       cdst = buf0;
  for (int it = 0; it < 10; ++it) {
    k_conv<<<(N_ROWS * (N_ROWS / 4)) / 256, 256, 0, stream>>>(
        csrc, cdst, rowsum, it == 0 ? 1 : 0,
        conv1_k, conv1_b, conv2_k, conv2_b, prelu_a);
    csrc = cdst;
    cdst = (cdst == buf0) ? buf1 : buf0;
  }

  k_adjo<<<(N_ROWS * 32) / 256, 256, 0, stream>>>(csrc, scores, rowsum, inp,
                                                  g_pre);

  k_tail<<<N_ROWS / 256, 256, 0, stream>>>(g_pre, Wg, bg, gcn_a, Wp1, bp1,
                                           Wp2, bp2, (float*)d_out);
}
